// SimpleChannelAttn_7198365188194
// MI455X (gfx1250) — compile-verified, matrix operand order checked by reading, not run on hardware
//
#include <hip/hip_runtime.h>
#include <hip/hip_bf16.h>

// ---------------------------------------------------------------------------
// SimpleChannelAttn on MI455X (gfx1250, wave32, WMMA bf16)
// out = x + softmax(Q K^T / sqrt(512)) V, per-batch channel attention.
// All GEMMs via v_wmma_f32_16x16x32_bf16, 128x128 blocks, double-buffered LDS.
// ---------------------------------------------------------------------------

typedef __bf16 bf16;
typedef __attribute__((ext_vector_type(8)))  __bf16 v8bf;
typedef __attribute__((ext_vector_type(16))) __bf16 v16bf;
typedef __attribute__((ext_vector_type(8)))  float  v8f;

#define NB 32
#define CC 512
#define DD 1024
#define MM (NB * CC)               // 16384 total token rows
#define SCALE 0.04419417382415922f // 1/sqrt(512)

// ---------------------------------------------------------------------------
// f32 -> bf16 cast (vectorized)
// ---------------------------------------------------------------------------
__global__ __launch_bounds__(256) void cast_f32_to_bf16(
    const float* __restrict__ s, bf16* __restrict__ d, int n) {
  int i = (blockIdx.x * 256 + threadIdx.x) * 4;
  if (i + 4 <= n) {
    float4 v = *(const float4*)(s + i);
    d[i + 0] = (bf16)v.x;
    d[i + 1] = (bf16)v.y;
    d[i + 2] = (bf16)v.z;
    d[i + 3] = (bf16)v.w;
  }
}

// ---------------------------------------------------------------------------
// Fused cast + transpose of W (1024x1024 f32) -> Wt (1024x1024 bf16),
// Wt[n][k] = W[k][n]. LDS tile transpose, padded to dodge bank conflicts.
// ---------------------------------------------------------------------------
__global__ __launch_bounds__(256) void cast_transpose_w(
    const float* __restrict__ W0, const float* __restrict__ W1,
    const float* __restrict__ W2, bf16* __restrict__ Wt) {
  __shared__ bf16 t[64][66];
  const float* W = (blockIdx.z == 0) ? W0 : (blockIdx.z == 1 ? W1 : W2);
  bf16* O = Wt + (size_t)blockIdx.z * DD * DD;
  const int kBase = blockIdx.x * 64;
  const int nBase = blockIdx.y * 64;
#pragma unroll
  for (int i = 0; i < 16; ++i) {
    int idx = i * 256 + threadIdx.x;
    int r = idx >> 6, c = idx & 63;
    t[r][c] = (bf16)W[(size_t)(kBase + r) * DD + nBase + c];
  }
  __syncthreads();
#pragma unroll
  for (int i = 0; i < 16; ++i) {
    int idx = i * 256 + threadIdx.x;
    int r = idx >> 6, c = idx & 63;  // r = n-local, c = k-local
    O[(size_t)(nBase + r) * DD + kBase + c] = t[c][r];
  }
}

// ---------------------------------------------------------------------------
// Fragment load: 16x32 bf16 tile from LDS (row stride 32 elems).
// CDNA5 16-bit A/B layout: lane<16 -> row base+(lane&15), K in {0..7,16..23};
// lane>=16 -> same rows, K in {8..15,24..31}. Two aligned 16B ds reads.
// ---------------------------------------------------------------------------
__device__ __forceinline__ v16bf load_frag(const bf16* lds, int rowBase, int lane) {
  int r  = rowBase + (lane & 15);
  int kh = (lane >> 4) << 3;  // 0 or 8
  v8bf lo = *(const v8bf*)(lds + r * 32 + kh);
  v8bf hi = *(const v8bf*)(lds + r * 32 + 16 + kh);
  v16bf f;
#pragma unroll
  for (int j = 0; j < 8; ++j) { f[j] = lo[j]; f[j + 8] = hi[j]; }
  return f;
}

// ---------------------------------------------------------------------------
// Staging: 128x32 bf16 tile, 256 threads, 2 contiguous 16B chunks per thread.
// ---------------------------------------------------------------------------
__device__ __forceinline__ void stage_load(const bf16* __restrict__ src,
                                           int rowBase, int ld, int kb,
                                           int tid, v8bf reg[2]) {
#pragma unroll
  for (int i = 0; i < 2; ++i) {
    int c2 = tid * 2 + i;        // 0..511
    int r  = c2 >> 2;            // 0..127
    int cc = (c2 & 3) << 3;      // 0,8,16,24
    reg[i] = *(const v8bf*)(src + (size_t)(rowBase + r) * ld + kb + cc);
  }
}

__device__ __forceinline__ void stage_store(bf16* lds, int tid, const v8bf reg[2]) {
#pragma unroll
  for (int i = 0; i < 2; ++i) {
    int c2 = tid * 2 + i;
    int r  = c2 >> 2;
    int cc = (c2 & 3) << 3;
    *(v8bf*)(lds + r * 32 + cc) = reg[i];
  }
}

// Shared GEMM inner loop macro pieces are written out per kernel for clarity.

// ---------------------------------------------------------------------------
// QKV projection: for g in {0,1,2}: O_g = E(16384x1024) @ W_g.
// B comes from pre-transposed Wt (row = n, contiguous k). Q,K stored normal
// (bf16, MxD). V stored TRANSPOSED per batch: Vt[b][n][c] — the accumulator
// layout makes this a contiguous 16B store per tile.
// Block = 128x128, 256 threads (8 waves), wave = 32x64 (2x4 WMMA tiles).
// ---------------------------------------------------------------------------
__global__ __launch_bounds__(256) void gemm_qkv(
    const bf16* __restrict__ E, const bf16* __restrict__ Wt,
    bf16* __restrict__ Qo, bf16* __restrict__ Ko, bf16* __restrict__ Vt) {
  __shared__ __align__(16) bf16 As[2][128 * 32];
  __shared__ __align__(16) bf16 Bs[2][128 * 32];

  const int g = blockIdx.z;
  const bf16* Wg = Wt + (size_t)g * DD * DD;

  const int mBase = blockIdx.x * 128;
  const int nBase = blockIdx.y * 128;
  const int tid  = threadIdx.x;
  const int lane = tid & 31;
  const int wave = tid >> 5;
  const int wr = (wave >> 1) * 32;  // 0,32,64,96
  const int wc = (wave & 1) * 64;   // 0,64

  v8f zero = {};
  v8f acc[2][4];
#pragma unroll
  for (int i = 0; i < 2; ++i)
#pragma unroll
    for (int j = 0; j < 4; ++j) acc[i][j] = zero;

  v8bf aReg[2], bReg[2];
  stage_load(E,  mBase, DD, 0, tid, aReg);
  stage_load(Wg, nBase, DD, 0, tid, bReg);
  stage_store(As[0], tid, aReg);
  stage_store(Bs[0], tid, bReg);
  __syncthreads();

  const int nIter = DD / 32;
  for (int it = 0; it < nIter; ++it) {
    const int cur = it & 1;
    if (it + 1 < nIter) {
      stage_load(E,  mBase, DD, (it + 1) * 32, tid, aReg);
      stage_load(Wg, nBase, DD, (it + 1) * 32, tid, bReg);
    }
    v16bf af[2], bfr[4];
    af[0] = load_frag(As[cur], wr + 0,  lane);
    af[1] = load_frag(As[cur], wr + 16, lane);
#pragma unroll
    for (int j = 0; j < 4; ++j) bfr[j] = load_frag(Bs[cur], wc + j * 16, lane);
#pragma unroll
    for (int i = 0; i < 2; ++i)
#pragma unroll
      for (int j = 0; j < 4; ++j)
        acc[i][j] = __builtin_amdgcn_wmma_f32_16x16x32_bf16(
            false, af[i], false, bfr[j], (short)0, acc[i][j], false, false);
    if (it + 1 < nIter) {
      stage_store(As[cur ^ 1], tid, aReg);
      stage_store(Bs[cur ^ 1], tid, bReg);
    }
    __syncthreads();
  }

  const int nOff = lane & 15;
  const int mOff = (lane >> 4) * 8;
  if (g < 2) {
    bf16* O = (g == 0) ? Qo : Ko;
#pragma unroll
    for (int ti = 0; ti < 2; ++ti)
#pragma unroll
      for (int tj = 0; tj < 4; ++tj)
#pragma unroll
        for (int v = 0; v < 8; ++v) {
          int m = mBase + wr + ti * 16 + mOff + v;
          int n = nBase + wc + tj * 16 + nOff;
          O[(size_t)m * DD + n] = (bf16)acc[ti][tj][v];
        }
  } else {
    // V transposed: Vt[b][n][c]; 8 consecutive m (=c) per lane -> one 16B store
#pragma unroll
    for (int ti = 0; ti < 2; ++ti)
#pragma unroll
      for (int tj = 0; tj < 4; ++tj) {
        int m0 = mBase + wr + ti * 16 + mOff;
        int n  = nBase + wc + tj * 16 + nOff;
        int b  = m0 >> 9;          // /512
        int c0 = m0 & 511;
        v8bf pk;
#pragma unroll
        for (int v = 0; v < 8; ++v) pk[v] = (bf16)acc[ti][tj][v];
        *(v8bf*)(Vt + ((size_t)b * DD + n) * CC + c0) = pk;
      }
  }
}

// ---------------------------------------------------------------------------
// scores(b) = Q_b(512x1024) @ K_b^T * SCALE, f32 out. Both operands stage
// row-major (B columns are K-rows). Block 128x128, double-buffered.
// ---------------------------------------------------------------------------
__global__ __launch_bounds__(256) void gemm_scores(
    const bf16* __restrict__ Q, const bf16* __restrict__ Kmat,
    float* __restrict__ S) {
  __shared__ __align__(16) bf16 As[2][128 * 32];
  __shared__ __align__(16) bf16 Bs[2][128 * 32];

  const int b = blockIdx.z;
  const bf16* q = Q    + (size_t)b * CC * DD;
  const bf16* k = Kmat + (size_t)b * CC * DD;
  float* s = S + (size_t)b * CC * CC;

  const int mBase = blockIdx.x * 128;
  const int nBase = blockIdx.y * 128;
  const int tid  = threadIdx.x;
  const int lane = tid & 31;
  const int wave = tid >> 5;
  const int wr = (wave >> 1) * 32;
  const int wc = (wave & 1) * 64;

  v8f zero = {};
  v8f acc[2][4];
#pragma unroll
  for (int i = 0; i < 2; ++i)
#pragma unroll
    for (int j = 0; j < 4; ++j) acc[i][j] = zero;

  v8bf aReg[2], bReg[2];
  stage_load(q, mBase, DD, 0, tid, aReg);
  stage_load(k, nBase, DD, 0, tid, bReg);
  stage_store(As[0], tid, aReg);
  stage_store(Bs[0], tid, bReg);
  __syncthreads();

  const int nIter = DD / 32;
  for (int it = 0; it < nIter; ++it) {
    const int cur = it & 1;
    if (it + 1 < nIter) {
      stage_load(q, mBase, DD, (it + 1) * 32, tid, aReg);
      stage_load(k, nBase, DD, (it + 1) * 32, tid, bReg);
    }
    v16bf af[2], bfr[4];
    af[0] = load_frag(As[cur], wr + 0,  lane);
    af[1] = load_frag(As[cur], wr + 16, lane);
#pragma unroll
    for (int j = 0; j < 4; ++j) bfr[j] = load_frag(Bs[cur], wc + j * 16, lane);
#pragma unroll
    for (int i = 0; i < 2; ++i)
#pragma unroll
      for (int j = 0; j < 4; ++j)
        acc[i][j] = __builtin_amdgcn_wmma_f32_16x16x32_bf16(
            false, af[i], false, bfr[j], (short)0, acc[i][j], false, false);
    if (it + 1 < nIter) {
      stage_store(As[cur ^ 1], tid, aReg);
      stage_store(Bs[cur ^ 1], tid, bReg);
    }
    __syncthreads();
  }

  const int nOff = lane & 15;
  const int mOff = (lane >> 4) * 8;
#pragma unroll
  for (int ti = 0; ti < 2; ++ti)
#pragma unroll
    for (int tj = 0; tj < 4; ++tj)
#pragma unroll
      for (int v = 0; v < 8; ++v) {
        int m = mBase + wr + ti * 16 + mOff + v;
        int n = nBase + wc + tj * 16 + nOff;
        s[(size_t)m * CC + n] = acc[ti][tj][v] * SCALE;
      }
}

// ---------------------------------------------------------------------------
// Row softmax over 512-wide rows; one wave32 per row. f32 in, bf16 out.
// ---------------------------------------------------------------------------
__global__ __launch_bounds__(256) void softmax_rows(
    const float* __restrict__ S, bf16* __restrict__ P) {
  const int lane = threadIdx.x & 31;
  const int wave = threadIdx.x >> 5;
  const int row  = blockIdx.x * 8 + wave;  // 16384 rows total
  const float* sr = S + (size_t)row * CC;

  float vals[16];
  float mx = -3.0e38f;
#pragma unroll
  for (int i = 0; i < 16; ++i) {
    vals[i] = sr[lane + i * 32];
    mx = fmaxf(mx, vals[i]);
  }
#pragma unroll
  for (int o = 16; o > 0; o >>= 1) mx = fmaxf(mx, __shfl_xor(mx, o, 32));

  float sum = 0.0f;
#pragma unroll
  for (int i = 0; i < 16; ++i) {
    vals[i] = __expf(vals[i] - mx);
    sum += vals[i];
  }
#pragma unroll
  for (int o = 16; o > 0; o >>= 1) sum += __shfl_xor(sum, o, 32);

  const float inv = 1.0f / sum;
  bf16* pr = P + (size_t)row * CC;
#pragma unroll
  for (int i = 0; i < 16; ++i) pr[lane + i * 32] = (bf16)(vals[i] * inv);
}

// ---------------------------------------------------------------------------
// out(b) = x(b) + attn_b(512x512) @ V_b. V comes pre-transposed (Vt[b][n][c]),
// so B stages with contiguous 16B loads. Fused skip-add epilogue, f32 out.
// ---------------------------------------------------------------------------
__global__ __launch_bounds__(256) void gemm_out(
    const bf16* __restrict__ P, const bf16* __restrict__ Vt,
    const float* __restrict__ X, float* __restrict__ Out) {
  __shared__ __align__(16) bf16 As[2][128 * 32];
  __shared__ __align__(16) bf16 Bs[2][128 * 32];

  const int b = blockIdx.z;
  const bf16* p  = P  + (size_t)b * CC * CC;
  const bf16* vt = Vt + (size_t)b * DD * CC;  // rows = n (1024), cols = k (512)
  const float* xo = X + (size_t)b * CC * DD;
  float* oo = Out + (size_t)b * CC * DD;

  const int mBase = blockIdx.x * 128;  // rows in [0,512)
  const int nBase = blockIdx.y * 128;  // cols in [0,1024)
  const int tid  = threadIdx.x;
  const int lane = tid & 31;
  const int wave = tid >> 5;
  const int wr = (wave >> 1) * 32;
  const int wc = (wave & 1) * 64;

  v8f zero = {};
  v8f acc[2][4];
#pragma unroll
  for (int i = 0; i < 2; ++i)
#pragma unroll
    for (int j = 0; j < 4; ++j) acc[i][j] = zero;

  v8bf aReg[2], bReg[2];
  stage_load(p,  mBase, CC, 0, tid, aReg);
  stage_load(vt, nBase, CC, 0, tid, bReg);
  stage_store(As[0], tid, aReg);
  stage_store(Bs[0], tid, bReg);
  __syncthreads();

  const int nIter = CC / 32;
  for (int it = 0; it < nIter; ++it) {
    const int cur = it & 1;
    if (it + 1 < nIter) {
      stage_load(p,  mBase, CC, (it + 1) * 32, tid, aReg);
      stage_load(vt, nBase, CC, (it + 1) * 32, tid, bReg);
    }
    v16bf af[2], bfr[4];
    af[0] = load_frag(As[cur], wr + 0,  lane);
    af[1] = load_frag(As[cur], wr + 16, lane);
#pragma unroll
    for (int j = 0; j < 4; ++j) bfr[j] = load_frag(Bs[cur], wc + j * 16, lane);
#pragma unroll
    for (int i = 0; i < 2; ++i)
#pragma unroll
      for (int j = 0; j < 4; ++j)
        acc[i][j] = __builtin_amdgcn_wmma_f32_16x16x32_bf16(
            false, af[i], false, bfr[j], (short)0, acc[i][j], false, false);
    if (it + 1 < nIter) {
      stage_store(As[cur ^ 1], tid, aReg);
      stage_store(Bs[cur ^ 1], tid, bReg);
    }
    __syncthreads();
  }

  const int nOff = lane & 15;
  const int mOff = (lane >> 4) * 8;
#pragma unroll
  for (int ti = 0; ti < 2; ++ti)
#pragma unroll
    for (int tj = 0; tj < 4; ++tj)
#pragma unroll
      for (int vv = 0; vv < 8; ++vv) {
        int m = mBase + wr + ti * 16 + mOff + vv;
        int n = nBase + wc + tj * 16 + nOff;
        size_t idx = (size_t)m * DD + n;
        oo[idx] = xo[idx] + acc[ti][tj][vv];
      }
}

// ---------------------------------------------------------------------------
// Workspace layout (bytes):
//   e      @ 0          : 16384*1024 bf16   = 32 MiB
//   Wt     @ 33554432   : 3*1024*1024 bf16  =  6 MiB (transposed)
//   q      @ 39845888   : 16384*1024 bf16   = 32 MiB
//   k      @ 73400320   : 16384*1024 bf16   = 32 MiB
//   vt     @ 106954752  : 32*1024*512 bf16  = 32 MiB (per-batch transposed)
//   scores @ 140509184  : 32*512*512 f32    = 32 MiB
//   attn   @ 174063616  : 32*512*512 bf16   = 16 MiB
// ---------------------------------------------------------------------------
extern "C" void kernel_launch(void* const* d_in, const int* in_sizes, int n_in,
                              void* d_out, int out_size, void* d_ws, size_t ws_size,
                              hipStream_t stream) {
  const float* x  = (const float*)d_in[0];
  const float* Wq = (const float*)d_in[1];
  const float* Wk = (const float*)d_in[2];
  const float* Wv = (const float*)d_in[3];
  float* out = (float*)d_out;

  char* ws = (char*)d_ws;
  bf16*  e   = (bf16*)(ws + 0);
  bf16*  wt  = (bf16*)(ws + 33554432);
  bf16*  q   = (bf16*)(ws + 39845888);
  bf16*  k   = (bf16*)(ws + 73400320);
  bf16*  vt  = (bf16*)(ws + 106954752);
  float* sc  = (float*)(ws + 140509184);
  bf16*  at  = (bf16*)(ws + 174063616);

  // 1) cast x -> bf16; cast+transpose weights -> Wt
  cast_f32_to_bf16<<<16384, 256, 0, stream>>>(x, e, MM * DD);
  cast_transpose_w<<<dim3(16, 16, 3), 256, 0, stream>>>(Wq, Wk, Wv, wt);

  // 2) Q, K (normal), Vt (transposed) = E @ {Wq,Wk,Wv}
  gemm_qkv<<<dim3(MM / 128, DD / 128, 3), 256, 0, stream>>>(e, wt, q, k, vt);

  // 3) scores = Q K^T / sqrt(C)
  gemm_scores<<<dim3(CC / 128, CC / 128, NB), 256, 0, stream>>>(q, k, sc);

  // 4) row softmax -> bf16 attention
  softmax_rows<<<MM / 8, 256, 0, stream>>>(sc, at);

  // 5) out = x + attn @ V
  gemm_out<<<dim3(CC / 128, DD / 128, NB), 256, 0, stream>>>(at, vt, x, out);
}
